// MLPPredictor_9689446219935
// MI455X (gfx1250) — matrix operand order repaired and checked by reading, not compiled, not run on hardware
//
#include <hip/hip_runtime.h>

// CDNA5 / gfx1250 edge-MLP scorer using V_WMMA_F32_16X16X32_F16 (wave32).
//
//   he = [h[src] | h[dst]]  (E x 256, f32 -> f16 on the fly)
//   z  = relu(he @ W1^T + b1)            layer 1: WMMA GEMM
//   s  = z @ W2^T + b2                   layer 2: VALU + shfl reduction
//
// Each wave processes TWO groups of 16 edges so every B fragment fetched
// from LDS feeds two v_wmma instructions (halves LDS read bandwidth).

typedef __attribute__((ext_vector_type(16))) _Float16 v16h;
typedef __attribute__((ext_vector_type(8)))  float    v8f;

#define HF  128            // hidden feats (layer-1 output width)
#define KF  256            // 2*HF (layer-1 input width)
#define NT  8              // N tiles  (128 / 16)
#define KT  8              // K tiles  (256 / 32)
#define EPG 16             // edges per group (WMMA M)
#define THREADS 256
#define WAVES (THREADS / 32)

__global__ __launch_bounds__(THREADS)
void edge_mlp_wmma(const float* __restrict__ h,
                   const int*   __restrict__ src,
                   const int*   __restrict__ dst,
                   const float* __restrict__ W1,   // [128][256] row-major: W1[j][k]
                   const float* __restrict__ b1,   // [128]
                   const float* __restrict__ W2,   // [128]
                   const float* __restrict__ b2,   // [1]
                   float*       __restrict__ out,  // [E]
                   int E, int npairs)
{
    // W1^T staged as f16 WMMA B-fragments: B[k][n] = W1[n][k].
    // Fragment (kt,nt), lane l, element i:
    //   n = nt*16 + (l & 15),  k = kt*32 + (l >> 4)*16 + i   (i: 16 consecutive k)
    __shared__ v16h ldsB[KT][NT][32];   // 64 KB

    for (int slot = threadIdx.x; slot < KT * NT * 32; slot += THREADS) {
        int l  = slot & 31;
        int nt = (slot >> 5) & (NT - 1);
        int kt =  slot >> 8;
        int n  = nt * 16 + (l & 15);
        int k  = kt * 32 + (l >> 4) * 16;
        const float* wp = W1 + n * KF + k;      // 16 consecutive floats
        v16h frag;
        #pragma unroll
        for (int i = 0; i < 16; ++i) frag[i] = (_Float16)wp[i];
        ldsB[kt][nt][l] = frag;                 // 2x ds_store_b128
    }
    __syncthreads();

    const int lane = threadIdx.x & 31;
    const int half = lane >> 4;   // which 16-lane half of the wave
    const int m    = lane & 15;   // edge row within a group (WMMA M for A)

    // Epilogue constants, per lane (lane maps to N within an N-tile).
    float b1v[NT], w2v[NT];
    #pragma unroll
    for (int nt = 0; nt < NT; ++nt) {
        b1v[nt] = b1[nt * 16 + m];
        w2v[nt] = W2[nt * 16 + m];
    }
    const float b2s = b2[0];

    const int wave = threadIdx.x >> 5;
    for (int p = blockIdx.x * WAVES + wave; p < npairs; p += gridDim.x * WAVES) {
        // Opaque memory barrier: stops LICM from hoisting the (loop-invariant)
        // LDS B-fragment loads out of this loop, which previously forced a
        // 2KB/lane spill to scratch. Keeps them as in-loop ds_load_b128.
        asm volatile("" ::: "memory");

        const int base0 = (2 * p)     * EPG;
        const int base1 = (2 * p + 1) * EPG;
        int e0 = base0 + m; if (e0 >= E) e0 = E - 1;   // tail clamp
        int e1 = base1 + m; if (e1 >= E) e1 = E - 1;
        const float* srow0 = h + (size_t)src[e0] * HF;
        const float* drow0 = h + (size_t)dst[e0] * HF;
        const float* srow1 = h + (size_t)src[e1] * HF;
        const float* drow1 = h + (size_t)dst[e1] * HF;

        v8f acc0[NT], acc1[NT];
        #pragma unroll
        for (int nt = 0; nt < NT; ++nt) { v8f z = {}; acc0[nt] = z; acc1[nt] = z; }

        #pragma unroll
        for (int kt = 0; kt < KT; ++kt) {
            // A fragment 16x32 f16: VGPR v holds K = half*16 + {2v,2v+1}.
            const float* rp0 = (kt < 4) ? srow0 : drow0;
            const float* rp1 = (kt < 4) ? srow1 : drow1;
            const int kb = (kt & 3) * 32 + half * 16;
            v16h a0, a1;
            #pragma unroll
            for (int v = 0; v < 8; ++v) {
                int kl = kb + 2 * v;
                float2 f0 = *(const float2*)(rp0 + kl);   // L2-resident gather
                float2 f1 = *(const float2*)(rp1 + kl);
                a0[2 * v]     = (_Float16)f0.x;
                a0[2 * v + 1] = (_Float16)f0.y;
                a1[2 * v]     = (_Float16)f1.x;
                a1[2 * v + 1] = (_Float16)f1.y;
            }
            #pragma unroll
            for (int nt = 0; nt < NT; ++nt) {
                v16h bfrag = ldsB[kt][nt][lane];          // 2x ds_load_b128, reused 2x
                acc0[nt] = __builtin_amdgcn_wmma_f32_16x16x32_f16(
                    false, a0, false, bfrag, (short)0, acc0[nt], false, false);
                acc1[nt] = __builtin_amdgcn_wmma_f32_16x16x32_f16(
                    false, a1, false, bfrag, (short)0, acc1[nt], false, false);
            }
        }

        // Layer 2: score[m'] = sum_n relu(Z[m'][n] + b1[n]) * W2[n] + b2
        // C/D layout: VGPR r -> M=r (lanes 0-15) / M=r+8 (lanes 16-31), lane -> N.
        #pragma unroll
        for (int grp = 0; grp < 2; ++grp) {
            const v8f* acc = grp ? acc1 : acc0;
            const int  bse = grp ? base1 : base0;
            float t[8];
            #pragma unroll
            for (int r = 0; r < 8; ++r) t[r] = 0.0f;
            #pragma unroll
            for (int nt = 0; nt < NT; ++nt) {
                #pragma unroll
                for (int r = 0; r < 8; ++r) {
                    float z = acc[nt][r] + b1v[nt];
                    z = z > 0.0f ? z : 0.0f;
                    t[r] += z * w2v[nt];
                }
            }
            #pragma unroll
            for (int r = 0; r < 8; ++r) {   // reduce over 16 lanes of each half
                t[r] += __shfl_xor(t[r], 1, 32);
                t[r] += __shfl_xor(t[r], 2, 32);
                t[r] += __shfl_xor(t[r], 4, 32);
                t[r] += __shfl_xor(t[r], 8, 32);
            }
            if (m == 0) {                   // lanes 0 and 16 each own 8 edges
                #pragma unroll
                for (int r = 0; r < 8; ++r) {
                    int eo = bse + half * 8 + r;
                    if (eo < E) out[eo] = t[r] + b2s;
                }
            }
        }
    }
}

extern "C" void kernel_launch(void* const* d_in, const int* in_sizes, int n_in,
                              void* d_out, int out_size, void* d_ws, size_t ws_size,
                              hipStream_t stream) {
    const float* h   = (const float*)d_in[0];
    const int*   src = (const int*)  d_in[1];
    const int*   dst = (const int*)  d_in[2];
    const float* W1  = (const float*)d_in[3];
    const float* b1  = (const float*)d_in[4];
    const float* W2  = (const float*)d_in[5];
    const float* b2  = (const float*)d_in[6];
    float* out = (float*)d_out;

    const int E = in_sizes[1];                   // 600000 edges
    const int ngroups = (E + EPG - 1) / EPG;     // 37500 groups of 16 edges
    const int npairs  = (ngroups + 1) / 2;       // 18750 group-pairs (one per wave iter)

    int blocks = (npairs + WAVES - 1) / WAVES;
    if (blocks > 512) blocks = 512;              // ~4-5 pairs per wave; amortizes staging
    edge_mlp_wmma<<<blocks, THREADS, 0, stream>>>(h, src, dst, W1, b1, W2, b2,
                                                  out, E, npairs);
}
